// RNNModel_11742440587672
// MI455X (gfx1250) — matrix operand order repaired and checked by reading, not compiled, not run on hardware
//
#include <hip/hip_runtime.h>
#include <hip/hip_bf16.h>

#define RNN_B 64
#define RNN_T 512
#define RNN_D 512
#define RNN_H 1024

typedef __attribute__((ext_vector_type(16))) __bf16 v16bf;
typedef __attribute__((ext_vector_type(8)))  __bf16 v8bf;
typedef __attribute__((ext_vector_type(8)))  float  v8f;

// ---- workspace layout (bytes) ----
static constexpr size_t XW_OFF   = 0;                                           // 128 MiB fp32 [T][B][H]
static constexpr size_t WHT_OFF  = XW_OFF  + (size_t)RNN_T * RNN_B * RNN_H * 4; // 2 MiB bf16 WhT[n][k]
static constexpr size_t WXT_OFF  = WHT_OFF + (size_t)RNN_H * RNN_H * 2;         // 1 MiB bf16 WxT[n][k]
static constexpr size_t H0_OFF   = WXT_OFF + (size_t)RNN_H * RNN_D * 2;         // 128 KiB bf16 h buf 0
static constexpr size_t H1_OFF   = H0_OFF  + (size_t)RNN_B * RNN_H * 2;         // 128 KiB bf16 h buf 1
static constexpr size_t SYNC_OFF = H1_OFF  + (size_t)RNN_B * RNN_H * 2;         // 2 u32

__device__ __forceinline__ __bf16 f2bf(float f) { return (__bf16)f; }

// ---------------- prep: transpose+convert weights, zero h0, reset barrier ----
__global__ void k_prep(const float* __restrict__ Wx, const float* __restrict__ Wh,
                       __bf16* __restrict__ WxT, __bf16* __restrict__ WhT,
                       __bf16* __restrict__ h0, unsigned* __restrict__ sync) {
  const size_t idx = (size_t)blockIdx.x * blockDim.x + threadIdx.x;
  if (idx < (size_t)RNN_H * RNN_H) {           // Wh[k][n] -> WhT[n][k]
    const int k = (int)(idx / RNN_H), n = (int)(idx % RNN_H);
    WhT[(size_t)n * RNN_H + k] = f2bf(Wh[idx]);
  }
  if (idx < (size_t)RNN_D * RNN_H) {           // Wx[k][n] -> WxT[n][k]
    const int k = (int)(idx / RNN_H), n = (int)(idx % RNN_H);
    WxT[(size_t)n * RNN_D + k] = f2bf(Wx[idx]);
  }
  if (idx < (size_t)RNN_B * RNN_H) h0[idx] = f2bf(0.0f);
  if (idx < 2) sync[idx] = 0u;
}

// ---------------- phase 1: xw = x @ Wx + b, stored time-major [T][B][H] -----
// One wave computes a 16x128 strip: 1 M-tile x 8 N-tiles (A fragment reused 8x).
__global__ void k_xw_gemm(const float* __restrict__ x, const __bf16* __restrict__ WxT,
                          const float* __restrict__ bias, float* __restrict__ xw) {
  const int wave = blockIdx.x * (blockDim.x >> 5) + (threadIdx.x >> 5);
  const int lane = threadIdx.x & 31;
  const int ng = wave & 7;         // N-group: 8 groups of 8 tiles = H/16 tiles
  const int mt = wave >> 3;        // M-tile: (B*T)/16 = 2048 tiles
  const int r16  = lane & 15;
  const int half = lane >> 4;

  const float*  pa0 = x + (size_t)(mt * 16 + r16) * RNN_D + half * 16;
  const __bf16* pb0 = WxT + (size_t)((ng * 8) * 16 + r16) * RNN_D + half * 16;

  v8f acc[8] = {};
  for (int kk = 0; kk < RNN_D; kk += 32) {
    // preload all 8 B fragments
    v16bf bfrag[8];
#pragma unroll
    for (int nt = 0; nt < 8; ++nt)
      bfrag[nt] = *(const v16bf*)(pb0 + (size_t)nt * 16 * RNN_D + kk);

    // A fragment: lane holds row (lane%16), K = half*16 + {0..15}
    const float* pa = pa0 + kk;
    v16bf a;
#pragma unroll
    for (int j = 0; j < 8; ++j) {
      a[j]     = f2bf(pa[j]);
      a[j + 8] = f2bf(pa[j + 8]);
    }
#pragma unroll
    for (int nt = 0; nt < 8; ++nt)
      acc[nt] = __builtin_amdgcn_wmma_f32_16x16x32_bf16(
          false, a, false, bfrag[nt], (short)0, acc[nt], false, false);
  }
  // epilogue: add bias, scatter to time-major xw
#pragma unroll
  for (int nt = 0; nt < 8; ++nt) {
    const int n  = (ng * 8 + nt) * 16 + r16;
    const float bv = bias[n];
#pragma unroll
    for (int r = 0; r < 8; ++r) {
      const int mrow = mt * 16 + r + half * 8;     // m = b*T + t
      const int t  = mrow & (RNN_T - 1);
      const int bb = mrow >> 9;                    // /T
      xw[(size_t)t * RNN_B * RNN_H + (size_t)bb * RNN_H + n] = acc[nt][r] + bv;
    }
  }
}

// ---------------- device-wide barrier (64 co-resident blocks) ---------------
__device__ __forceinline__ void grid_barrier(unsigned* cnt, unsigned* gen, unsigned nblk) {
  __threadfence();
  __syncthreads();
  if (threadIdx.x == 0) {
    unsigned g = __hip_atomic_load(gen, __ATOMIC_ACQUIRE, __HIP_MEMORY_SCOPE_AGENT);
    unsigned prev = atomicAdd(cnt, 1u);
    if (prev == nblk - 1u) {
      *cnt = 0u;
      __hip_atomic_store(gen, g + 1u, __ATOMIC_RELEASE, __HIP_MEMORY_SCOPE_AGENT);
    } else {
      while (__hip_atomic_load(gen, __ATOMIC_ACQUIRE, __HIP_MEMORY_SCOPE_AGENT) == g) {
        __builtin_amdgcn_s_sleep(1);
      }
    }
  }
  __syncthreads();
  __threadfence();
}

// ---------------- phase 2: persistent recurrent scan (LDS-staged) -----------
// 64 blocks x 4 waves. Block b: M-tile = b>>4, N-tiles = (b&15)*4 .. +3 (one per wave).
// Wh tiles (128 KiB) staged in LDS once for all 512 steps; the 32 KiB h M-tile
// is cooperatively staged in LDS each step so K-loop loads are ds_load_b128.
// K-loop is FULLY unrolled: 32 WMMAs + 128 ds_loads straight-line, so the LDS
// loads issue far ahead of the dependent WMMA accumulation chain.
__global__ void k_rnn_scan(const float* __restrict__ xw, const __bf16* __restrict__ WhT,
                           __bf16* __restrict__ h0, __bf16* __restrict__ h1,
                           unsigned* __restrict__ sync) {
  extern __shared__ __bf16 smem[];
  __bf16* sWh = smem;                         // [64 local cols][H] = 65536 elems (128 KiB)
  __bf16* sA  = smem + 64 * RNN_H;            // [16 rows][H]       = 16384 elems (32 KiB)

  const int tid  = threadIdx.x;               // 0..127
  const int w    = tid >> 5;                  // wave 0..3 -> local N-tile
  const int lane = tid & 31;
  const int mt = blockIdx.x >> 4;             // 0..3   (batch tile)
  const int ng = blockIdx.x & 15;             // 0..15  (N-group of 4 tiles)
  const int r16  = lane & 15;
  const int half = lane >> 4;
  const int ncol = (ng * 4 + w) * 16 + r16;   // global output column this lane owns

  // ---- stage this block's 64 Wh columns into LDS (contiguous 128 KiB) ----
  {
    const __bf16* src = WhT + (size_t)(ng * 64) * RNN_H;
    for (int i = tid * 8; i < 64 * RNN_H; i += 128 * 8)
      *(v8bf*)(&sWh[i]) = *(const v8bf*)(src + i);
  }
  __syncthreads();

  const __bf16* pb0 = sWh + (size_t)(w * 16 + r16) * RNN_H + half * 16;  // LDS B base
  const __bf16* pa0 = sA + (size_t)r16 * RNN_H + half * 16;              // LDS A base

  __bf16* hb[2] = { h0, h1 };

  for (int s = 0; s < RNN_T; ++s) {
    const __bf16* hp = hb[s & 1];
    __bf16*       hn = hb[(s + 1) & 1];

    // ---- cooperatively stage h M-tile (16 x H bf16 = 32 KiB) into LDS ----
    {
      const __bf16* src = hp + (size_t)(mt * 16) * RNN_H;
      for (int i = tid * 8; i < 16 * RNN_H; i += 128 * 8)
        *(v8bf*)(&sA[i]) = *(const v8bf*)(src + i);
    }
    // prefetch the xw values this lane's epilogue will read
    const float* xwt = xw + (size_t)s * RNN_B * RNN_H;
#pragma unroll
    for (int r = 0; r < 8; ++r)
      __builtin_prefetch(xwt + (size_t)(mt * 16 + r + half * 8) * RNN_H + ncol, 0, 3);
    __syncthreads();

    v8f acc = {};
#pragma unroll
    for (int kk = 0; kk < RNN_H; kk += 32) {
      const v8bf  lo = *(const v8bf*)(pa0 + kk);
      const v8bf  hi = *(const v8bf*)(pa0 + kk + 8);
      const v16bf b  = *(const v16bf*)(pb0 + kk);
      v16bf a;
#pragma unroll
      for (int j = 0; j < 8; ++j) { a[j] = lo[j]; a[j + 8] = hi[j]; }
      acc = __builtin_amdgcn_wmma_f32_16x16x32_bf16(
          false, a, false, b, (short)0, acc, false, false);
    }
    // epilogue: h_new = tanh(xw[t] + h@Wh), requantize to bf16
#pragma unroll
    for (int r = 0; r < 8; ++r) {
      const int brow = mt * 16 + r + half * 8;
      const size_t o = (size_t)brow * RNN_H + ncol;
      hn[o] = f2bf(tanhf(acc[r] + xwt[o]));
    }
    grid_barrier(sync, sync + 1, gridDim.x);   // also fences sA reuse next step
  }
}

// ---------------- phase 3: out = h_last @ Wfc + bfc -------------------------
__global__ void k_fc(const __bf16* __restrict__ h, const float* __restrict__ Wfc,
                     const float* __restrict__ bfc, float* __restrict__ out) {
  const int b = threadIdx.x;
  if (b < RNN_B) {
    float s = 0.0f;
    for (int k = 0; k < RNN_H; ++k) s += (float)h[(size_t)b * RNN_H + k] * Wfc[k];
    out[b] = s + bfc[0];
  }
}

extern "C" void kernel_launch(void* const* d_in, const int* in_sizes, int n_in,
                              void* d_out, int out_size, void* d_ws, size_t ws_size,
                              hipStream_t stream) {
  const float* x    = (const float*)d_in[0];
  const float* Wx   = (const float*)d_in[1];
  const float* Wh   = (const float*)d_in[2];
  const float* bias = (const float*)d_in[3];
  const float* Wfc  = (const float*)d_in[4];
  const float* bfc  = (const float*)d_in[5];
  float* out = (float*)d_out;

  char* ws = (char*)d_ws;
  float*    xw  = (float*)(ws + XW_OFF);
  __bf16*   WhT = (__bf16*)(ws + WHT_OFF);
  __bf16*   WxT = (__bf16*)(ws + WXT_OFF);
  __bf16*   h0  = (__bf16*)(ws + H0_OFF);
  __bf16*   h1  = (__bf16*)(ws + H1_OFF);
  unsigned* syn = (unsigned*)(ws + SYNC_OFF);

  // prep: H*H = 1M elements / 256 threads
  k_prep<<<dim3((RNN_H * RNN_H + 255) / 256), dim3(256), 0, stream>>>(
      Wx, Wh, WxT, WhT, h0, syn);

  // phase 1: 2048 M-tiles * 8 N-groups = 16384 waves / 4 per block
  k_xw_gemm<<<dim3(16384 / 4), dim3(128), 0, stream>>>(x, WxT, bias, xw);

  // phase 2: persistent, 64 blocks * 4 waves; 160 KiB dynamic LDS per block
  const size_t shmem = (size_t)(64 * RNN_H + 16 * RNN_H) * sizeof(__bf16);
  k_rnn_scan<<<dim3(64), dim3(128), shmem, stream>>>(xw, WhT, h0, h1, syn);

  // phase 3: h_last lives in h0 after 512 steps
  k_fc<<<dim3(1), dim3(64), 0, stream>>>(h0, Wfc, bfc, out);
}
